// TapeHead_1185410974179
// MI455X (gfx1250) — matrix operand order repaired and checked by reading, not compiled, not run on hardware
//
#include <hip/hip_runtime.h>
#include <math.h>

// ---------------------------------------------------------------------------
// TapeHead transformer block for MI455X (gfx1250), wave32 + WMMA bf16.
// B=4, S=2048, D=1024, C=2 (window), H=16, dh=64.
// All GEMMs + attention use v_wmma_f32_16x16x32_bf16 (f32 accumulate).
// Software-pipelined global->reg->LDS staging, 16B-aligned LDS strides so
// fragment loads merge to ds_load_b128.
// ---------------------------------------------------------------------------

#define BQ 4
#define SQ 2048
#define DQ 1024
#define HQ 16
#define DH 64

// LDS strides (dwords). 20 and 36 are multiples of 4 dwords (16B) and give a
// conflict-free 16-lane spread (gcd(20,64)=gcd(36,64)=4 -> 16 distinct banks).
#define LDSW 20
#define LDKW 36

typedef __attribute__((ext_vector_type(16))) __bf16 v16bf;
typedef __attribute__((ext_vector_type(8)))  float  v8f;

union Frag16 { v16bf v; unsigned int u[8]; };

__device__ __forceinline__ unsigned short f2bf(float x) {
  unsigned int u = __float_as_uint(x);
  return (unsigned short)((u + 0x7FFFu + ((u >> 16) & 1u)) >> 16);
}

// ---------------------------------------------------------------------------
// Elementwise f32 -> bf16 cast, 4 elements per thread
// ---------------------------------------------------------------------------
__global__ void __launch_bounds__(256) cast_bf16x4_k(const float4* __restrict__ src,
                                                     uint2* __restrict__ dst,
                                                     int n4) {
  int i = blockIdx.x * 256 + threadIdx.x;
  if (i < n4) {
    float4 f = src[i];
    uint2 o;
    o.x = (unsigned int)f2bf(f.x) | ((unsigned int)f2bf(f.y) << 16);
    o.y = (unsigned int)f2bf(f.z) | ((unsigned int)f2bf(f.w) << 16);
    dst[i] = o;
  }
}

// ---------------------------------------------------------------------------
// Shared GEMM geometry: 256 threads = 8 waves (2x4), block tile 128(M)x256(N),
// wave tile 64x64 (4 M-frags x 4 N-frags), K-step 32 halves.
// A tile: 128x32 bf16 in LDS (stride LDSW dwords). B tile: 256x32 (N-major).
// ---------------------------------------------------------------------------

// GEMM 1: chunk = windows @ chunk_w^T + chunk_b
__global__ void __launch_bounds__(256) chunk_gemm_k(
    const unsigned int* __restrict__ xb,      // bf16 (B,S,D) as dwords
    const unsigned int* __restrict__ wb,      // bf16 (D, 2D) as dwords
    const float* __restrict__ bias,           // (D)
    float* __restrict__ chunk_f,
    unsigned short* __restrict__ chunk_bf) {
  __shared__ unsigned int lA[128 * LDSW];
  __shared__ unsigned int lB[256 * LDSW];
  const int K = 2 * DQ;
  int tid = threadIdx.x;
  int lane = tid & 31, wid = tid >> 5;
  int halfL = lane >> 4, l16 = lane & 15;
  int bm = blockIdx.x * 128, bn = blockIdx.y * 256;
  int wm = (wid >> 2) * 64, wn = (wid & 3) * 64;

  int arow = tid >> 3, acd = tid & 7;        // A stage: 4 x (row+=32)
  int brow = tid >> 3, bcd = tid & 7;        // B stage: 8 x (row+=32)

  v8f acc[4][4];
#pragma unroll
  for (int i = 0; i < 4; ++i)
#pragma unroll
    for (int nf = 0; nf < 4; ++nf)
#pragma unroll
      for (int j = 0; j < 8; ++j) acc[i][nf][j] = 0.f;

  uint2 ra[4], rb[8];
  auto load_tiles = [&](int kt) {
    int hi = (kt >= DQ) ? 1 : 0;             // wave-uniform window select
    int kb = kt - hi * DQ;
#pragma unroll
    for (int i = 0; i < 4; ++i) {
      int row = arow + i * 32;
      int m = bm + row;
      int bb_ = m >> 11, ss_ = (m & (SQ - 1)) + hi;
      uint2 v; v.x = 0u; v.y = 0u;
      if (ss_ < SQ)
        v = ((const uint2*)xb)[((size_t)(bb_ * SQ + ss_) * DQ + kb + acd * 4) >> 2];
      ra[i] = v;
    }
#pragma unroll
    for (int i = 0; i < 8; ++i) {
      int row = brow + i * 32;
      rb[i] = ((const uint2*)wb)[((size_t)(bn + row) * K + kt + bcd * 4) >> 2];
    }
  };
  auto store_tiles = [&]() {
#pragma unroll
    for (int i = 0; i < 4; ++i)
      ((uint2*)(lA + (arow + i * 32) * LDSW))[acd] = ra[i];
#pragma unroll
    for (int i = 0; i < 8; ++i)
      ((uint2*)(lB + (brow + i * 32) * LDSW))[bcd] = rb[i];
  };

  load_tiles(0);
  for (int kt = 0; kt < K; kt += 32) {
    __syncthreads();
    store_tiles();
    __syncthreads();
    if (kt + 32 < K) load_tiles(kt + 32);    // prefetch next tile (latency hide)

    Frag16 af[4];
#pragma unroll
    for (int i = 0; i < 4; ++i) {
      int row = wm + i * 16 + l16;
#pragma unroll
      for (int d = 0; d < 8; ++d) {
        int kdw = halfL * 8 + d;
        af[i].u[d] = lA[row * LDSW + kdw];
      }
    }
#pragma unroll
    for (int nf = 0; nf < 4; ++nf) {
      Frag16 bf;
      int n = wn + nf * 16 + l16;
#pragma unroll
      for (int d = 0; d < 8; ++d) bf.u[d] = lB[n * LDSW + halfL * 8 + d];
#pragma unroll
      for (int i = 0; i < 4; ++i)
        acc[i][nf] = __builtin_amdgcn_wmma_f32_16x16x32_bf16(
            false, af[i].v, false, bf.v, (short)0, acc[i][nf], false, false);
    }
  }

#pragma unroll
  for (int i = 0; i < 4; ++i)
#pragma unroll
    for (int nf = 0; nf < 4; ++nf) {
      int col = bn + wn + nf * 16 + l16;
      float bv = bias[col];
#pragma unroll
      for (int j = 0; j < 8; ++j) {
        int m = bm + wm + i * 16 + j + halfL * 8;
        float v = acc[i][nf][j] + bv;
        chunk_f[(size_t)m * DQ + col] = v;
        chunk_bf[(size_t)m * DQ + col] = f2bf(v);
      }
    }
}

// GEMM 2: qkv = chunk @ in_proj_w^T + in_proj_b, scattered to (part,B,H,S,dh)
__global__ void __launch_bounds__(256) qkv_gemm_k(
    const unsigned int* __restrict__ ab,      // chunk bf16 (B*S, D) as dwords
    const unsigned int* __restrict__ wb,      // in_proj_w bf16 (3D, D)
    const float* __restrict__ bias,           // (3D)
    unsigned short* __restrict__ qkv) {       // bf16 (3, B, H, S, dh)
  __shared__ unsigned int lA[128 * LDSW];
  __shared__ unsigned int lB[256 * LDSW];
  const int K = DQ;
  int tid = threadIdx.x;
  int lane = tid & 31, wid = tid >> 5;
  int halfL = lane >> 4, l16 = lane & 15;
  int bm = blockIdx.x * 128, bn = blockIdx.y * 256;
  int wm = (wid >> 2) * 64, wn = (wid & 3) * 64;
  int arow = tid >> 3, acd = tid & 7;

  v8f acc[4][4];
#pragma unroll
  for (int i = 0; i < 4; ++i)
#pragma unroll
    for (int nf = 0; nf < 4; ++nf)
#pragma unroll
      for (int j = 0; j < 8; ++j) acc[i][nf][j] = 0.f;

  uint2 ra[4], rb[8];
  auto load_tiles = [&](int kt) {
#pragma unroll
    for (int i = 0; i < 4; ++i)
      ra[i] = ((const uint2*)ab)[((size_t)(bm + arow + i * 32) * DQ + kt + acd * 4) >> 2];
#pragma unroll
    for (int i = 0; i < 8; ++i)
      rb[i] = ((const uint2*)wb)[((size_t)(bn + arow + i * 32) * K + kt + acd * 4) >> 2];
  };
  auto store_tiles = [&]() {
#pragma unroll
    for (int i = 0; i < 4; ++i)
      ((uint2*)(lA + (arow + i * 32) * LDSW))[acd] = ra[i];
#pragma unroll
    for (int i = 0; i < 8; ++i)
      ((uint2*)(lB + (arow + i * 32) * LDSW))[acd] = rb[i];
  };

  load_tiles(0);
  for (int kt = 0; kt < K; kt += 32) {
    __syncthreads();
    store_tiles();
    __syncthreads();
    if (kt + 32 < K) load_tiles(kt + 32);

    Frag16 af[4];
#pragma unroll
    for (int i = 0; i < 4; ++i) {
      int row = wm + i * 16 + l16;
#pragma unroll
      for (int d = 0; d < 8; ++d) {
        int kdw = halfL * 8 + d;
        af[i].u[d] = lA[row * LDSW + kdw];
      }
    }
#pragma unroll
    for (int nf = 0; nf < 4; ++nf) {
      Frag16 bf;
      int n = wn + nf * 16 + l16;
#pragma unroll
      for (int d = 0; d < 8; ++d) bf.u[d] = lB[n * LDSW + halfL * 8 + d];
#pragma unroll
      for (int i = 0; i < 4; ++i)
        acc[i][nf] = __builtin_amdgcn_wmma_f32_16x16x32_bf16(
            false, af[i].v, false, bf.v, (short)0, acc[i][nf], false, false);
    }
  }

#pragma unroll
  for (int i = 0; i < 4; ++i)
#pragma unroll
    for (int nf = 0; nf < 4; ++nf) {
      int e = bn + wn + nf * 16 + l16;
      float bv = bias[e];
      int part = e >> 10;
      int hh = (e & 1023) >> 6;
      int dd = e & 63;
#pragma unroll
      for (int j = 0; j < 8; ++j) {
        int m = bm + wm + i * 16 + j + halfL * 8;
        int bb = m >> 11, ss = m & (SQ - 1);
        size_t o = ((((size_t)part * BQ + bb) * HQ + hh) * SQ + ss) * DH + dd;
        qkv[o] = f2bf(acc[i][nf][j] + bv);
      }
    }
}

// ---------------------------------------------------------------------------
// Flash attention: per (b,h), 8 waves x 16 query rows per block, key chunks
// of 32 staged in LDS (prefetched into regs), online softmax, P routed
// C-layout -> A-layout via per-wave LDS scratch.
// ---------------------------------------------------------------------------
__global__ void __launch_bounds__(256) attn_k(
    const unsigned int* __restrict__ qkv,     // bf16 (3,B,H,S,dh) as dwords
    unsigned short* __restrict__ attn_bf) {   // bf16 (B,S,D)
  __shared__ unsigned int  lK[32 * LDKW];      // [key][dh/2]
  __shared__ unsigned short lV[64 * 2 * LDSW]; // [dh][key], stride 40 halves
  __shared__ unsigned short lP[8][16 * 2 * LDSW]; // per-wave P scratch

  const size_t PART = (size_t)BQ * HQ * SQ * (DH / 2);  // dwords per part
  int tid = threadIdx.x, lane = tid & 31, wid = tid >> 5;
  int halfL = lane >> 4, l16 = lane & 15;
  int bb = blockIdx.z, hh = blockIdx.y;
  int q0 = blockIdx.x * 128 + wid * 16;
  size_t base = (size_t)(bb * HQ + hh) * SQ * (DH / 2);  // dwords
  const unsigned int* qh = qkv;
  const uint2* kh2 = (const uint2*)(qkv + PART);
  const uint2* vh2 = (const uint2*)(qkv + 2 * PART);
  size_t base2 = base >> 1;                              // uint2 units

  // Q: two resident A fragments (dh 0..31, 32..63)
  Frag16 A0, A1;
  {
    int qrow = q0 + l16;
#pragma unroll
    for (int d = 0; d < 8; ++d) {
      int kdw = halfL * 8 + d;
      A0.u[d] = qh[base + (size_t)qrow * 32 + kdw];
      A1.u[d] = qh[base + (size_t)qrow * 32 + 16 + kdw];
    }
  }

  float m_[8], l_[8];
  v8f acc[4];
#pragma unroll
  for (int j = 0; j < 8; ++j) { m_[j] = -3.0e38f; l_[j] = 0.f; }
#pragma unroll
  for (int nb = 0; nb < 4; ++nb)
#pragma unroll
    for (int j = 0; j < 8; ++j) acc[nb][j] = 0.f;

  const float scale = 0.125f;  // 1/sqrt(64)
  int crow = tid >> 4, ccd = tid & 15;       // stage: 2 x (row+=16), 16 uint2/row

  uint2 rk[2], rv[2];
  auto load_kv = [&](int kc) {
#pragma unroll
    for (int i = 0; i < 2; ++i) {
      int row = crow + i * 16;
      rk[i] = kh2[base2 + (size_t)(kc + row) * 16 + ccd];
      rv[i] = vh2[base2 + (size_t)(kc + row) * 16 + ccd];
    }
  };
  auto store_kv = [&]() {
#pragma unroll
    for (int i = 0; i < 2; ++i) {
      int row = crow + i * 16;
      ((uint2*)(lK + row * LDKW))[ccd] = rk[i];
      // V transposed: halves dh = 4*ccd .. 4*ccd+3 of this key row
      lV[(4 * ccd + 0) * (2 * LDSW) + row] = (unsigned short)(rv[i].x & 0xFFFFu);
      lV[(4 * ccd + 1) * (2 * LDSW) + row] = (unsigned short)(rv[i].x >> 16);
      lV[(4 * ccd + 2) * (2 * LDSW) + row] = (unsigned short)(rv[i].y & 0xFFFFu);
      lV[(4 * ccd + 3) * (2 * LDSW) + row] = (unsigned short)(rv[i].y >> 16);
    }
  };

  load_kv(0);
  for (int kc = 0; kc < SQ; kc += 32) {
    __syncthreads();
    store_kv();
    __syncthreads();
    if (kc + 32 < SQ) load_kv(kc + 32);      // prefetch next K/V chunk

    // scores: two 16x16 tiles (keys kc..+15, kc+16..+31)
    float p0[8], p1[8];
#pragma unroll
    for (int kk = 0; kk < 2; ++kk) {
      v8f s;
#pragma unroll
      for (int j = 0; j < 8; ++j) s[j] = 0.f;
      int key = kk * 16 + l16;
#pragma unroll
      for (int ks = 0; ks < 2; ++ks) {
        Frag16 bfr;
#pragma unroll
        for (int d = 0; d < 8; ++d)
          bfr.u[d] = lK[key * LDKW + ks * 16 + halfL * 8 + d];
        s = __builtin_amdgcn_wmma_f32_16x16x32_bf16(
            false, (ks ? A1.v : A0.v), false, bfr.v, (short)0, s, false, false);
      }
#pragma unroll
      for (int j = 0; j < 8; ++j) {
        float v = s[j] * scale;
        if (kk) p1[j] = v; else p0[j] = v;
      }
    }

    // online softmax update (per-row stats within each 16-lane group)
    float corr[8];
#pragma unroll
    for (int j = 0; j < 8; ++j) {
      float v = fmaxf(p0[j], p1[j]);
      v = fmaxf(v, __shfl_xor(v, 8, 32));
      v = fmaxf(v, __shfl_xor(v, 4, 32));
      v = fmaxf(v, __shfl_xor(v, 2, 32));
      v = fmaxf(v, __shfl_xor(v, 1, 32));
      float mn = fmaxf(m_[j], v);
      corr[j] = __expf(m_[j] - mn);
      m_[j] = mn;
    }
#pragma unroll
    for (int j = 0; j < 8; ++j) {
      p0[j] = __expf(p0[j] - m_[j]);
      p1[j] = __expf(p1[j] - m_[j]);
      int prow = j + halfL * 8;
      lP[wid][prow * (2 * LDSW) + l16] = f2bf(p0[j]);
      lP[wid][prow * (2 * LDSW) + 16 + l16] = f2bf(p1[j]);
      float sum = p0[j] + p1[j];
      sum += __shfl_xor(sum, 8, 32);
      sum += __shfl_xor(sum, 4, 32);
      sum += __shfl_xor(sum, 2, 32);
      sum += __shfl_xor(sum, 1, 32);
      l_[j] = l_[j] * corr[j] + sum;
    }
#pragma unroll
    for (int nb = 0; nb < 4; ++nb)
#pragma unroll
      for (int j = 0; j < 8; ++j) acc[nb][j] *= corr[j];

    // wave-local LDS ordering: P stores must land before P frag loads
    asm volatile("s_wait_dscnt 0" ::: "memory");

    Frag16 pf;
    {
      const unsigned int* lPu = (const unsigned int*)(&lP[wid][0]);
#pragma unroll
      for (int d = 0; d < 8; ++d) {
        int kdw = halfL * 8 + d;
        pf.u[d] = lPu[l16 * LDSW + kdw];
      }
    }
    const unsigned int* lVu = (const unsigned int*)lV;
#pragma unroll
    for (int nb = 0; nb < 4; ++nb) {
      Frag16 vf;
      int dh = nb * 16 + l16;
#pragma unroll
      for (int d = 0; d < 8; ++d) vf.u[d] = lVu[dh * LDSW + halfL * 8 + d];
      acc[nb] = __builtin_amdgcn_wmma_f32_16x16x32_bf16(
          false, pf.v, false, vf.v, (short)0, acc[nb], false, false);
    }
  }

  // epilogue: normalize and scatter to (B,S,D) with D index = h*64+dh
#pragma unroll
  for (int nb = 0; nb < 4; ++nb)
#pragma unroll
    for (int j = 0; j < 8; ++j) {
      int row = q0 + j + halfL * 8;
      int dh = nb * 16 + l16;
      float v = acc[nb][j] / l_[j];
      attn_bf[(size_t)(bb * SQ + row) * DQ + hh * DH + dh] = f2bf(v);
    }
}

// GEMM 3: resid = chunk_f + attn @ out_proj_w^T + out_proj_b
__global__ void __launch_bounds__(256) out_gemm_k(
    const unsigned int* __restrict__ ab,      // attn bf16 (B*S, D) as dwords
    const unsigned int* __restrict__ wb,      // out_proj_w bf16 (D, D)
    const float* __restrict__ bias,           // (D)
    const float* __restrict__ chunk_f,
    float* __restrict__ resid) {
  __shared__ unsigned int lA[128 * LDSW];
  __shared__ unsigned int lB[256 * LDSW];
  const int K = DQ;
  int tid = threadIdx.x;
  int lane = tid & 31, wid = tid >> 5;
  int halfL = lane >> 4, l16 = lane & 15;
  int bm = blockIdx.x * 128, bn = blockIdx.y * 256;
  int wm = (wid >> 2) * 64, wn = (wid & 3) * 64;
  int arow = tid >> 3, acd = tid & 7;

  v8f acc[4][4];
#pragma unroll
  for (int i = 0; i < 4; ++i)
#pragma unroll
    for (int nf = 0; nf < 4; ++nf)
#pragma unroll
      for (int j = 0; j < 8; ++j) acc[i][nf][j] = 0.f;

  uint2 ra[4], rb[8];
  auto load_tiles = [&](int kt) {
#pragma unroll
    for (int i = 0; i < 4; ++i)
      ra[i] = ((const uint2*)ab)[((size_t)(bm + arow + i * 32) * DQ + kt + acd * 4) >> 2];
#pragma unroll
    for (int i = 0; i < 8; ++i)
      rb[i] = ((const uint2*)wb)[((size_t)(bn + arow + i * 32) * K + kt + acd * 4) >> 2];
  };
  auto store_tiles = [&]() {
#pragma unroll
    for (int i = 0; i < 4; ++i)
      ((uint2*)(lA + (arow + i * 32) * LDSW))[acd] = ra[i];
#pragma unroll
    for (int i = 0; i < 8; ++i)
      ((uint2*)(lB + (arow + i * 32) * LDSW))[acd] = rb[i];
  };

  load_tiles(0);
  for (int kt = 0; kt < K; kt += 32) {
    __syncthreads();
    store_tiles();
    __syncthreads();
    if (kt + 32 < K) load_tiles(kt + 32);

    Frag16 af[4];
#pragma unroll
    for (int i = 0; i < 4; ++i) {
      int row = wm + i * 16 + l16;
#pragma unroll
      for (int d = 0; d < 8; ++d) {
        int kdw = halfL * 8 + d;
        af[i].u[d] = lA[row * LDSW + kdw];
      }
    }
#pragma unroll
    for (int nf = 0; nf < 4; ++nf) {
      Frag16 bf;
      int n = wn + nf * 16 + l16;
#pragma unroll
      for (int d = 0; d < 8; ++d) bf.u[d] = lB[n * LDSW + halfL * 8 + d];
#pragma unroll
      for (int i = 0; i < 4; ++i)
        acc[i][nf] = __builtin_amdgcn_wmma_f32_16x16x32_bf16(
            false, af[i].v, false, bf.v, (short)0, acc[i][nf], false, false);
    }
  }

#pragma unroll
  for (int i = 0; i < 4; ++i)
#pragma unroll
    for (int nf = 0; nf < 4; ++nf) {
      int col = bn + wn + nf * 16 + l16;
      float bv = bias[col];
#pragma unroll
      for (int j = 0; j < 8; ++j) {
        int m = bm + wm + i * 16 + j + halfL * 8;
        size_t o = (size_t)m * DQ + col;
        resid[o] = acc[i][nf][j] + bv + chunk_f[o];
      }
    }
}

// ---------------------------------------------------------------------------
// LayerNorm over rows of 1024, one block per row
// ---------------------------------------------------------------------------
__global__ void __launch_bounds__(256) ln_k(const float* __restrict__ resid,
                                            const float* __restrict__ g,
                                            const float* __restrict__ bta,
                                            float* __restrict__ out) {
  __shared__ float red0[8], red1[8];
  __shared__ float smu, srs;
  int row = blockIdx.x, tid = threadIdx.x;
  const float* x = resid + (size_t)row * DQ;
  float v[4], s = 0.f, s2 = 0.f;
#pragma unroll
  for (int i = 0; i < 4; ++i) {
    v[i] = x[tid + i * 256];
    s += v[i];
    s2 += v[i] * v[i];
  }
#pragma unroll
  for (int m = 16; m; m >>= 1) {
    s += __shfl_xor(s, m, 32);
    s2 += __shfl_xor(s2, m, 32);
  }
  int wid = tid >> 5, lane = tid & 31;
  if (lane == 0) { red0[wid] = s; red1[wid] = s2; }
  __syncthreads();
  if (tid == 0) {
    float t = 0.f, t2 = 0.f;
#pragma unroll
    for (int i = 0; i < 8; ++i) { t += red0[i]; t2 += red1[i]; }
    float mu = t * (1.f / DQ);
    float var = t2 * (1.f / DQ) - mu * mu;
    smu = mu;
    srs = rsqrtf(var + 1e-5f);
  }
  __syncthreads();
  float mu = smu, rs = srs;
#pragma unroll
  for (int i = 0; i < 4; ++i) {
    int c = tid + i * 256;
    out[(size_t)row * DQ + c] = (v[i] - mu) * rs * g[c] + bta[c];
  }
}

// ---------------------------------------------------------------------------
// Host orchestration. Workspace layout (~172 MB):
//   xb, w_chunk, w_in, w_out (bf16), chunk_f (f32), chunk_bf, qkv (bf16),
//   attn_bf (bf16), resid (f32)
// ---------------------------------------------------------------------------
extern "C" void kernel_launch(void* const* d_in, const int* in_sizes, int n_in,
                              void* d_out, int out_size, void* d_ws, size_t ws_size,
                              hipStream_t stream) {
  (void)in_sizes; (void)n_in; (void)out_size; (void)ws_size;
  const float* x        = (const float*)d_in[0];
  const float* chunk_w  = (const float*)d_in[1];
  const float* chunk_b  = (const float*)d_in[2];
  const float* in_w     = (const float*)d_in[3];
  const float* in_b     = (const float*)d_in[4];
  const float* out_w    = (const float*)d_in[5];
  const float* out_b    = (const float*)d_in[6];
  const float* ln_g     = (const float*)d_in[7];
  const float* ln_b     = (const float*)d_in[8];
  float* out = (float*)d_out;

  char* ws = (char*)d_ws;
  size_t off = 0;
  auto take = [&](size_t bytes) -> char* {
    char* p = ws + off;
    off += (bytes + 255) & ~(size_t)255;
    return p;
  };
  const size_t NX = (size_t)BQ * SQ * DQ;           // 8.4M elements
  unsigned short* xb       = (unsigned short*)take(NX * 2);
  unsigned short* wc_bf    = (unsigned short*)take((size_t)DQ * 2 * DQ * 2);
  unsigned short* wi_bf    = (unsigned short*)take((size_t)3 * DQ * DQ * 2);
  unsigned short* wo_bf    = (unsigned short*)take((size_t)DQ * DQ * 2);
  float*          chunk_f  = (float*)take(NX * 4);
  unsigned short* chunk_bf = (unsigned short*)take(NX * 2);
  unsigned short* qkv      = (unsigned short*)take(3 * NX * 2);
  unsigned short* attn_bf  = (unsigned short*)take(NX * 2);
  float*          resid    = (float*)take(NX * 4);

  // 1. casts (4 elements/thread)
  {
    int n4 = (int)(NX / 4);
    cast_bf16x4_k<<<(n4 + 255) / 256, 256, 0, stream>>>((const float4*)x, (uint2*)xb, n4);
    n4 = DQ * 2 * DQ / 4;
    cast_bf16x4_k<<<(n4 + 255) / 256, 256, 0, stream>>>((const float4*)chunk_w, (uint2*)wc_bf, n4);
    n4 = 3 * DQ * DQ / 4;
    cast_bf16x4_k<<<(n4 + 255) / 256, 256, 0, stream>>>((const float4*)in_w, (uint2*)wi_bf, n4);
    n4 = DQ * DQ / 4;
    cast_bf16x4_k<<<(n4 + 255) / 256, 256, 0, stream>>>((const float4*)out_w, (uint2*)wo_bf, n4);
  }
  // 2. chunk GEMM (M=8192, N=1024, K=2048)
  chunk_gemm_k<<<dim3(64, 4), 256, 0, stream>>>(
      (const unsigned int*)xb, (const unsigned int*)wc_bf, chunk_b,
      chunk_f, chunk_bf);
  // 3. qkv GEMM (M=8192, N=3072, K=1024)
  qkv_gemm_k<<<dim3(64, 12), 256, 0, stream>>>(
      (const unsigned int*)chunk_bf, (const unsigned int*)wi_bf, in_b, qkv);
  // 4. attention (S tiles x H x B)
  attn_k<<<dim3(SQ / 128, HQ, BQ), 256, 0, stream>>>(
      (const unsigned int*)qkv, attn_bf);
  // 5. out-proj GEMM + residual (M=8192, N=1024, K=1024)
  out_gemm_k<<<dim3(64, 4), 256, 0, stream>>>(
      (const unsigned int*)attn_bf, (const unsigned int*)wo_bf, out_b,
      chunk_f, resid);
  // 6. layernorm
  ln_k<<<BQ * SQ, 256, 0, stream>>>(resid, ln_g, ln_b, out);
}
